// CrossAttention_17136919511366
// MI455X (gfx1250) — matrix operand order repaired and checked by reading, not compiled, not run on hardware
//
#include <hip/hip_runtime.h>

// ---------------------------------------------------------------------------
// CDNA5 (gfx1250) cross-attention: B=2, S=2048, D=1024, H=16, DH=64
// wave32, 16x64 WMMA tile per wave in GEMMs (A-frag reuse x4),
// flash attention with online softmax:
//   - row-max via DPP row_ror VALU reduction (no DS traffic)
//   - row-sum via an extra WMMA against a ones B-fragment
// f16 WMMA / f32 accumulate throughout.
// ---------------------------------------------------------------------------

typedef __attribute__((ext_vector_type(16))) _Float16 v16h;
typedef __attribute__((ext_vector_type(8)))  _Float16 v8h;
typedef __attribute__((ext_vector_type(8)))  float    v8f;
typedef _Float16 half_t;

#define Bn  2
#define Sn  2048
#define Dn  1024
#define Hn  16
#define DHn 64

static __device__ __forceinline__ v8f wmma_f16(v16h a, v16h b, v8f c) {
  // D = A(16x32 f16) * B(32x16 f16) + C(16x16 f32)
  return __builtin_amdgcn_wmma_f32_16x16x32_f16(
      /*neg_a=*/false, a, /*neg_b=*/false, b,
      /*c_mod=*/(short)0, c, /*reuse_a=*/false, /*reuse_b=*/false);
}

// Max-reduce across each 16-lane half via DPP row_ror (VALU-only, co-executes
// with the WMMA pipe).  Rotation reduction: after steps 1,2,4,8 every lane in
// the 16-lane row holds the row maximum.
static __device__ __forceinline__ float rowmax16(float x) {
  int t;
  t = __builtin_amdgcn_update_dpp(0, __builtin_bit_cast(int, x),
                                  0x121 /*row_ror:1*/, 0xF, 0xF, true);
  x = fmaxf(x, __builtin_bit_cast(float, t));
  t = __builtin_amdgcn_update_dpp(0, __builtin_bit_cast(int, x),
                                  0x122 /*row_ror:2*/, 0xF, 0xF, true);
  x = fmaxf(x, __builtin_bit_cast(float, t));
  t = __builtin_amdgcn_update_dpp(0, __builtin_bit_cast(int, x),
                                  0x124 /*row_ror:4*/, 0xF, 0xF, true);
  x = fmaxf(x, __builtin_bit_cast(float, t));
  t = __builtin_amdgcn_update_dpp(0, __builtin_bit_cast(int, x),
                                  0x128 /*row_ror:8*/, 0xF, 0xF, true);
  x = fmaxf(x, __builtin_bit_cast(float, t));
  return x;
}

// A-fragment (16-bit 16x32): lane row = lane%16,
//   k(j) = j + 16*(lane>=16)  ->  two contiguous 8-half loads.
static __device__ __forceinline__ v16h load_afrag_h(const half_t* row, int lhi) {
  v8h lo = *(const v8h*)(row + 16 * lhi);
  v8h hi = *(const v8h*)(row + 8 + 16 * lhi);
  v16h r;
#pragma unroll
  for (int j = 0; j < 8; ++j) { r[j] = lo[j]; r[j + 8] = hi[j]; }
  return r;
}

// A-fragment from fp32 source (convert to f16 on the fly).
static __device__ __forceinline__ v16h load_afrag_f32(const float* row, int lhi) {
  const float4* p = (const float4*)(row + 16 * lhi);
  const float4* q = (const float4*)(row + 8 + 16 * lhi);
  float4 x0 = p[0], x1 = p[1], y0 = q[0], y1 = q[1];
  v16h a;
  a[0] = (half_t)x0.x; a[1] = (half_t)x0.y; a[2]  = (half_t)x0.z; a[3]  = (half_t)x0.w;
  a[4] = (half_t)x1.x; a[5] = (half_t)x1.y; a[6]  = (half_t)x1.z; a[7]  = (half_t)x1.w;
  a[8] = (half_t)y0.x; a[9] = (half_t)y0.y; a[10] = (half_t)y0.z; a[11] = (half_t)y0.w;
  a[12] = (half_t)y1.x; a[13] = (half_t)y1.y; a[14] = (half_t)y1.z; a[15] = (half_t)y1.w;
  return a;
}

// ---------------------------------------------------------------------------
// Kernel 0: transpose + f16-convert the 4 weight matrices (K x N -> N x K).
// Makes every WMMA B-fragment a single contiguous 32-byte global load.
// ---------------------------------------------------------------------------
__global__ __launch_bounds__(256) void transpose_weights_kernel(
    const float* __restrict__ Wq, const float* __restrict__ Wk,
    const float* __restrict__ Wv, const float* __restrict__ Wo,
    half_t* __restrict__ wqT, half_t* __restrict__ wkT,
    half_t* __restrict__ wvT, half_t* __restrict__ woT) {
  int idx = blockIdx.x * 256 + threadIdx.x;     // 0 .. 1M-1
  int sel = blockIdx.y;
  int n = idx >> 10, k = idx & 1023;
  const float* W = (sel == 0) ? Wq : (sel == 1) ? Wk : (sel == 2) ? Wv : Wo;
  half_t* WT     = (sel == 0) ? wqT : (sel == 1) ? wkT : (sel == 2) ? wvT : woT;
  WT[n * Dn + k] = (half_t)W[k * Dn + n];
}

// ---------------------------------------------------------------------------
// Kernel 1 (templated on SEL: 0=Q 1=K 2=V): X(4096x1024 f32) @ W -> f16.
// Each wave computes a 16x64 tile: one A-fragment reused across 4 B-frags,
// 4 f32 accumulators.  64 output columns == exactly one head.
// Q: [B,H,S,DH] scaled by DH^-0.5 ; K: [B,H,S,DH] ; V: [B,H,DH,S].
// ---------------------------------------------------------------------------
template <int SEL>
__global__ __launch_bounds__(128) void qkv_proj_kernel(
    const float* __restrict__ A, const half_t* __restrict__ WT,
    half_t* __restrict__ dst) {
  int lane = threadIdx.x & 31;
  int wave = threadIdx.x >> 5;
  int li = lane & 15, lhi = lane >> 4;
  int t = blockIdx.x * 4 + wave;                // 4096 tiles (256 M x 16 Ng)
  int ng = t & 15;                              // head / 64-col group
  int tileM = t >> 4;                           // 4096/16 rows

  const float*  arow = A + (size_t)(tileM * 16 + li) * Dn;
  const half_t* b0 = WT + (size_t)(ng * 64 + 0  * 16 + li) * Dn + 16 * lhi;
  const half_t* b1 = WT + (size_t)(ng * 64 + 1  * 16 + li) * Dn + 16 * lhi;
  const half_t* b2 = WT + (size_t)(ng * 64 + 2  * 16 + li) * Dn + 16 * lhi;
  const half_t* b3 = WT + (size_t)(ng * 64 + 3  * 16 + li) * Dn + 16 * lhi;

  v8f c0 = {}, c1 = {}, c2 = {}, c3 = {};
  for (int k0 = 0; k0 < Dn; k0 += 32) {
    __builtin_prefetch(arow + k0 + 32, 0, 1);   // global_prefetch_b8
    v16h a = load_afrag_f32(arow + k0, lhi);    // shared across 4 WMMAs
    c0 = wmma_f16(a, *(const v16h*)(b0 + k0), c0);
    c1 = wmma_f16(a, *(const v16h*)(b1 + k0), c1);
    c2 = wmma_f16(a, *(const v16h*)(b2 + k0), c2);
    c3 = wmma_f16(a, *(const v16h*)(b3 + k0), c3);
  }

  // store: head h = ng, d = nt*16 + li  (C-frag row = r + 8*lhi, col = li)
#pragma unroll
  for (int r = 0; r < 8; ++r) {
    int m = tileM * 16 + r + 8 * lhi;
    int bb = m >> 11, s = m & 2047;
    float v0 = c0[r], v1 = c1[r], v2 = c2[r], v3 = c3[r];
    if (SEL == 0) {                             // Q: pre-scaled by DH^-0.5
      half_t* p = dst + ((size_t)((bb * Hn + ng) * Sn + s)) * DHn + li;
      p[0]  = (half_t)(v0 * 0.125f);
      p[16] = (half_t)(v1 * 0.125f);
      p[32] = (half_t)(v2 * 0.125f);
      p[48] = (half_t)(v3 * 0.125f);
    } else if (SEL == 1) {                      // K: [B,H,S,DH]
      half_t* p = dst + ((size_t)((bb * Hn + ng) * Sn + s)) * DHn + li;
      p[0]  = (half_t)v0;
      p[16] = (half_t)v1;
      p[32] = (half_t)v2;
      p[48] = (half_t)v3;
    } else {                                    // V: transposed [B,H,DH,S]
      half_t* p = dst + ((size_t)((bb * Hn + ng) * DHn + li)) * Sn + s;
      p[(size_t)0 * Sn]  = (half_t)v0;
      p[(size_t)16 * Sn] = (half_t)v1;
      p[(size_t)32 * Sn] = (half_t)v2;
      p[(size_t)48 * Sn] = (half_t)v3;
    }
  }
}

// ---------------------------------------------------------------------------
// Kernel 2: flash attention. One wave = 16 query rows of one (b,h).
// Per 32-key block: 4 WMMAs for scores, DPP row-max, exp, P->LDS->A-frag,
// 1 WMMA for running row-sum (P @ ones), 4 WMMAs for P @ V.
// ---------------------------------------------------------------------------
__global__ __launch_bounds__(128) void attention_kernel(
    const half_t* __restrict__ Q, const half_t* __restrict__ K,
    const half_t* __restrict__ Vt, half_t* __restrict__ attn) {
  __shared__ half_t ldsP[4][16 * 32];           // per-wave P tile (16x32 f16)

  int lane = threadIdx.x & 31;
  int wave = threadIdx.x >> 5;
  int li = lane & 15, lhi = lane >> 4;
  int h = blockIdx.y, b = blockIdx.z;
  int qbase = (blockIdx.x * 4 + wave) * 16;

  const half_t* Qp = Q  + (size_t)((b * Hn + h) * Sn) * DHn;
  const half_t* Kp = K  + (size_t)((b * Hn + h) * Sn) * DHn;
  const half_t* Vp = Vt + (size_t)((b * Hn + h) * DHn) * Sn;  // [DH][S]

  // Q A-fragments for both K-steps over DH=64 (held in registers all loop).
  const half_t* qr = Qp + (size_t)(qbase + li) * DHn;
  v16h aq0 = load_afrag_h(qr, lhi);
  v16h aq1 = load_afrag_h(qr + 32, lhi);

  // ones B-fragment: P @ ones gives per-row sums in every C column.
  v16h ones;
#pragma unroll
  for (int j = 0; j < 16; ++j) ones[j] = (half_t)1.0f;

  float m[8];
#pragma unroll
  for (int r = 0; r < 8; ++r) m[r] = -__builtin_inff();
  v8f o0 = {}, o1 = {}, o2 = {}, o3 = {};
  v8f ls = {};                                  // running row-sum accumulator
  half_t* lp = ldsP[wave];

  for (int kb = 0; kb < Sn; kb += 32) {
    // ---- scores: 16 q-rows x 32 keys, contraction over DH=64 ----
    v8f sc0 = {}, sc1 = {};
    {
      const half_t* k0p = Kp + (size_t)(kb + li) * DHn + 16 * lhi;
      sc0 = wmma_f16(aq0, *(const v16h*)(k0p), sc0);
      sc0 = wmma_f16(aq1, *(const v16h*)(k0p + 32), sc0);
      const half_t* k1p = Kp + (size_t)(kb + 16 + li) * DHn + 16 * lhi;
      sc1 = wmma_f16(aq0, *(const v16h*)(k1p), sc1);
      sc1 = wmma_f16(aq1, *(const v16h*)(k1p + 32), sc1);
    }
    // ---- online softmax per row (rows r+8*lhi live in this 16-lane half) ----
#pragma unroll
    for (int r = 0; r < 8; ++r) {
      float v0 = sc0[r], v1 = sc1[r];
      float mx = rowmax16(fmaxf(v0, v1));        // DPP, VALU-only
      float mn = fmaxf(m[r], mx);
      float corr = __expf(m[r] - mn);            // 0 on first block
      m[r] = mn;
      float e0 = __expf(v0 - mn);
      float e1 = __expf(v1 - mn);
      o0[r] *= corr; o1[r] *= corr; o2[r] *= corr; o3[r] *= corr;
      ls[r] *= corr;
      int row = r + 8 * lhi;
      lp[row * 32 + li]      = (half_t)e0;       // C-layout -> LDS
      lp[row * 32 + 16 + li] = (half_t)e1;
    }
    // wave-private DS RAW hazard: our stores must land before A-frag reload
    asm volatile("s_wait_dscnt 0x0" ::: "memory");
    v16h ap = load_afrag_h(lp + li * 32, lhi);   // reload P in A-frag layout

    // ---- running row-sum: P(16x32) @ ones(32x16) ----
    ls = wmma_f16(ap, ones, ls);

    // ---- P(16x32) @ V(32x64): 4 N-tiles over DH, contiguous Vt loads ----
    const half_t* vp = Vp + (size_t)li * Sn + kb + 16 * lhi;
    o0 = wmma_f16(ap, *(const v16h*)(vp),            o0);
    o1 = wmma_f16(ap, *(const v16h*)(vp + 16 * Sn),  o1);
    o2 = wmma_f16(ap, *(const v16h*)(vp + 32 * Sn),  o2);
    o3 = wmma_f16(ap, *(const v16h*)(vp + 48 * Sn),  o3);
  }

  // ---- normalize and store to attn workspace [B,S,H*DH] f16 ----
#pragma unroll
  for (int r = 0; r < 8; ++r) {
    int q2 = qbase + r + 8 * lhi;
    float inv = 1.0f / ls[r];                    // row sum (same in all cols)
    size_t rowoff = ((size_t)(b * Sn + q2)) * Dn + h * DHn + li;
    attn[rowoff]      = (half_t)(o0[r] * inv);
    attn[rowoff + 16] = (half_t)(o1[r] * inv);
    attn[rowoff + 32] = (half_t)(o2[r] * inv);
    attn[rowoff + 48] = (half_t)(o3[r] * inv);
  }
}

// ---------------------------------------------------------------------------
// Kernel 3: output projection. attn(4096x1024 f16) @ Wo + bo -> f32 out.
// Same 16x64-per-wave tiling as the input projections.
// ---------------------------------------------------------------------------
__global__ __launch_bounds__(128) void out_proj_kernel(
    const half_t* __restrict__ attn, const half_t* __restrict__ woT,
    const float* __restrict__ bo, float* __restrict__ out) {
  int lane = threadIdx.x & 31;
  int wave = threadIdx.x >> 5;
  int li = lane & 15, lhi = lane >> 4;
  int t = blockIdx.x * 4 + wave;                // 4096 tiles
  int ng = t & 15;
  int tileM = t >> 4;

  const half_t* arow = attn + (size_t)(tileM * 16 + li) * Dn;
  const half_t* b0 = woT + (size_t)(ng * 64 + 0 * 16 + li) * Dn + 16 * lhi;
  const half_t* b1 = woT + (size_t)(ng * 64 + 1 * 16 + li) * Dn + 16 * lhi;
  const half_t* b2 = woT + (size_t)(ng * 64 + 2 * 16 + li) * Dn + 16 * lhi;
  const half_t* b3 = woT + (size_t)(ng * 64 + 3 * 16 + li) * Dn + 16 * lhi;

  v8f c0 = {}, c1 = {}, c2 = {}, c3 = {};
  for (int k0 = 0; k0 < Dn; k0 += 32) {
    __builtin_prefetch(arow + k0 + 32, 0, 1);
    v16h a = load_afrag_h(arow + k0, lhi);
    c0 = wmma_f16(a, *(const v16h*)(b0 + k0), c0);
    c1 = wmma_f16(a, *(const v16h*)(b1 + k0), c1);
    c2 = wmma_f16(a, *(const v16h*)(b2 + k0), c2);
    c3 = wmma_f16(a, *(const v16h*)(b3 + k0), c3);
  }

  int n = ng * 64 + li;
  float bias0 = bo[n], bias1 = bo[n + 16], bias2 = bo[n + 32], bias3 = bo[n + 48];
#pragma unroll
  for (int r = 0; r < 8; ++r) {
    int row = tileM * 16 + r + 8 * lhi;
    float* p = out + (size_t)row * Dn + n;
    p[0]  = c0[r] + bias0;
    p[16] = c1[r] + bias1;
    p[32] = c2[r] + bias2;
    p[48] = c3[r] + bias3;
  }
}

// ---------------------------------------------------------------------------
// Launch: ws layout (f16 elems): wqT|wkT|wvT|woT (1M each) then Q|K|Vt (4M
// each) then attn (4M).  Total 20M halves = 40 MB of d_ws.
// ---------------------------------------------------------------------------
extern "C" void kernel_launch(void* const* d_in, const int* in_sizes, int n_in,
                              void* d_out, int out_size, void* d_ws, size_t ws_size,
                              hipStream_t stream) {
  const float* query   = (const float*)d_in[0];
  const float* context = (const float*)d_in[1];
  const float* Wq      = (const float*)d_in[2];
  const float* Wk      = (const float*)d_in[3];
  const float* Wv      = (const float*)d_in[4];
  const float* Wo      = (const float*)d_in[5];
  const float* bo      = (const float*)d_in[6];

  half_t* ws   = (half_t*)d_ws;
  half_t* wqT  = ws;
  half_t* wkT  = wqT + (1u << 20);
  half_t* wvT  = wkT + (1u << 20);
  half_t* woT  = wvT + (1u << 20);
  half_t* Qh   = woT + (1u << 20);
  half_t* Kh   = Qh  + (4u << 20);
  half_t* Vh   = Kh  + (4u << 20);
  half_t* attn = Vh  + (4u << 20);

  transpose_weights_kernel<<<dim3(4096, 4), 256, 0, stream>>>(
      Wq, Wk, Wv, Wo, wqT, wkT, wvT, woT);
  qkv_proj_kernel<0><<<dim3(1024), 128, 0, stream>>>(query,   wqT, Qh);
  qkv_proj_kernel<1><<<dim3(1024), 128, 0, stream>>>(context, wkT, Kh);
  qkv_proj_kernel<2><<<dim3(1024), 128, 0, stream>>>(context, wvT, Vh);
  attention_kernel<<<dim3(32, 16, 2), 128, 0, stream>>>(Qh, Kh, Vh, attn);
  out_proj_kernel<<<dim3(4096), 128, 0, stream>>>(attn, woT, bo, (float*)d_out);
}
